// DKVMN_6107443495306
// MI455X (gfx1250) — matrix operand order repaired and checked by reading, not compiled, not run on hardware
//
#include <hip/hip_runtime.h>

// ---------------------------------------------------------------------------
// DKVMN forward, fused, for gfx1250 (MI455X).
//  - All GEMMs on v_wmma_f32_16x16x32_f16 (f16 in, f32 accumulate).
//  - Embed gather uses global_load_async_to_lds_b128 (ASYNCcnt path).
//  - B fragments reused across 4 row-tiles per wave (4x less global traffic
//    in the dominant 384x512 GEMM).
// Problem is compute-bound: ~113 GFLOP, <30MB hot data (L2-resident).
// ---------------------------------------------------------------------------

typedef __attribute__((ext_vector_type(16))) _Float16 v16h;
typedef __attribute__((ext_vector_type(4)))  _Float16 v4h;
typedef __attribute__((ext_vector_type(8)))  float    v8f;

#define THREADS   256
#define TILE_TOK  64
#define DIM_K     128
#define DIM_V     256
#define DIM_H     512
#define DIM_F     384          // V + K
#define MEM_PAD   64           // 50 memory slots padded to 64

#define FEAT_LD   392          // 384 + 8 halves pad (LDS bank stagger)
#define CORR_LD   72           // 64 + 8 halves pad

// workspace layout (in halves)
#define WS_WF16_OFF   0                                  // [512][384] f16
#define WS_KEY16_OFF  (DIM_H * DIM_F)                    // [64][128]  f16 (rows >=50 zero)
#define WS_VMT16_OFF  (WS_KEY16_OFF + MEM_PAD * DIM_K)   // [256][64]  f16 (V^T, k>=50 zero)

// LDS layout (bytes)
#define SM_FEAT    0
#define SM_LOGITS  (SM_FEAT   + TILE_TOK * FEAT_LD * 2)    // 50176
#define SM_CORR    (SM_LOGITS + TILE_TOK * MEM_PAD * 4)    // +16384
#define SM_BF      (SM_CORR   + TILE_TOK * CORR_LD * 2)    // +9216
#define SM_WP      (SM_BF     + DIM_H * 4)
#define SM_PRED    (SM_WP     + DIM_H * 4)
#define SM_STAGE   (SM_PRED   + TILE_TOK * 4)              // 80128
#define SM_BYTES   (SM_STAGE  + TILE_TOK * DIM_K * 4)      // +32768 = 112896

union HF16 { v16h v; float4 q[2]; };

// A-matrix fragment (16x32 f16) from LDS.  ISA layout: lane<16 -> row=lane,
// K{0..7,16..23}; lane>=16 -> row=lane-16, K{8..15,24..31}.
__device__ inline v16h lds_frag_a(const _Float16* rowbase, int ld, int lane) {
    const int r  = lane & 15;
    const int hi = lane >> 4;
    const _Float16* p = rowbase + r * ld + hi * 8;
    HF16 u;
    u.q[0] = *(const float4*)(p);
    u.q[1] = *(const float4*)(p + 16);
    return u.v;
}

// B-matrix fragment (32x16 f16) from global.  ISA layout: lane<16 -> col=lane,
// K{0..7,16..23}; lane>=16 -> col=lane-16, K{8..15,24..31}.
__device__ inline v16h glb_frag_b(const _Float16* colbase, int ld, int lane) {
    const int n  = lane & 15;
    const int hi = lane >> 4;
    const _Float16* p = colbase + n * ld + hi * 8;
    HF16 u;
    u.q[0] = *(const float4*)(p);
    u.q[1] = *(const float4*)(p + 16);
    return u.v;
}

__device__ inline v8f wmma_f16(v16h a, v16h b, v8f c) {
    return __builtin_amdgcn_wmma_f32_16x16x32_f16(
        /*neg_a=*/false, a, /*neg_b=*/false, b,
        /*c_mod=*/(short)0, c, /*reuse_a=*/false, /*reuse_b=*/false);
}

// ---------------------------------------------------------------------------
// Prep: f32 -> f16 weight conversion / padding / transpose into workspace.
// ---------------------------------------------------------------------------
__global__ __launch_bounds__(THREADS)
void dkvmn_prep(const float* __restrict__ W_f,       // [512][384]
                const float* __restrict__ key_mem,   // [50][128]
                const float* __restrict__ val_mem,   // [50][256]
                _Float16* __restrict__ ws) {
    int i = blockIdx.x * THREADS + threadIdx.x;
    const int NWF = DIM_H * DIM_F;
    if (i < NWF) { ws[WS_WF16_OFF + i] = (_Float16)W_f[i]; return; }
    i -= NWF;
    const int NK = MEM_PAD * DIM_K;
    if (i < NK) {
        int r = i >> 7, k = i & 127;
        ws[WS_KEY16_OFF + i] = (_Float16)(r < 50 ? key_mem[r * DIM_K + k] : 0.0f);
        return;
    }
    i -= NK;
    const int NV = DIM_V * MEM_PAD;
    if (i < NV) {
        int n = i >> 6, k = i & 63;   // vmT[n][k] = V[k][n]
        ws[WS_VMT16_OFF + i] = (_Float16)(k < 50 ? val_mem[k * DIM_V + n] : 0.0f);
    }
}

// ---------------------------------------------------------------------------
// Main fused kernel: 64 tokens / block, 256 threads (8 wave32s).
// ---------------------------------------------------------------------------
__global__ __launch_bounds__(THREADS)
void dkvmn_main(const int*   __restrict__ q,       // [B*S]
                const float* __restrict__ embed,   // [50001][128]
                const float* __restrict__ b_f,     // [512]
                const float* __restrict__ W_p,     // [512]
                const float* __restrict__ b_p,     // [1]
                const _Float16* __restrict__ ws,
                float* __restrict__ out) {
    extern __shared__ char smem[];
    _Float16* feat   = (_Float16*)(smem + SM_FEAT);
    float*    logits = (float*)   (smem + SM_LOGITS);
    _Float16* corr   = (_Float16*)(smem + SM_CORR);
    float*    bfs    = (float*)   (smem + SM_BF);
    float*    wps    = (float*)   (smem + SM_WP);
    float*    pacc   = (float*)   (smem + SM_PRED);
    float*    stage  = (float*)   (smem + SM_STAGE);   // [64][128] f32 embeds

    const _Float16* wf16  = ws + WS_WF16_OFF;
    const _Float16* key16 = ws + WS_KEY16_OFF;
    const _Float16* vmt16 = ws + WS_VMT16_OFF;

    const int tid  = threadIdx.x;
    const int lane = tid & 31;
    const int wave = tid >> 5;
    const int tok0 = blockIdx.x * TILE_TOK;

    // warm L2 with the W_f f16 slab this block will chew on
    __builtin_prefetch(wf16 + (size_t)tid * 64, 0, 0);

    // ---- phase 1a: async gather of embed rows (f32) straight into LDS.
    // Each lane issues b128 async loads: gather via per-lane global address,
    // per-lane LDS destination.  Tracked by ASYNCcnt.
    for (int i = tid; i < TILE_TOK * 32; i += THREADS) {   // 32 x b128 per token
        const int tok = i >> 5, c4 = i & 31;
        const int qi  = q[tok0 + tok];
        const int voff    = qi * (DIM_K * 4) + c4 * 16;    // byte offset in embed
        const int ldsaddr = SM_STAGE + tok * (DIM_K * 4) + c4 * 16;
        asm volatile("global_load_async_to_lds_b128 %0, %1, %2"
                     :: "v"(ldsaddr), "v"(voff), "s"(embed)
                     : "memory");
    }

    // constants + accumulator init while the gather flies
    for (int i = tid; i < DIM_H; i += THREADS) { bfs[i] = b_f[i]; wps[i] = W_p[i]; }
    if (tid < TILE_TOK) pacc[tid] = 0.0f;

    asm volatile("s_wait_asynccnt 0x0" ::: "memory");
    __syncthreads();

    // ---- phase 1b: convert staged f32 embeds -> f16 feat[:,256:384]
    for (int i = tid; i < TILE_TOK * 32; i += THREADS) {
        const int tok = i >> 5, c4 = i & 31;
        float4 v = *(const float4*)(stage + tok * DIM_K + c4 * 4);
        v4h h = { (_Float16)v.x, (_Float16)v.y, (_Float16)v.z, (_Float16)v.w };
        *(v4h*)(feat + tok * FEAT_LD + DIM_V + c4 * 4) = h;
    }
    __syncthreads();

    // ---- phase 2: logits[64x64] = qemb[64x128] @ key16^T
    // 16 tiles (4tr x 4tc); wave w: tc = w&3, tr in {(w>>2)*2, +1}; B reused.
    {
        const int tc  = wave & 3;
        const int trb = (wave >> 2) * 2;
        v8f acc0 = {}, acc1 = {};
        #pragma unroll
        for (int kk = 0; kk < 4; ++kk) {
            v16h b  = glb_frag_b(key16 + tc * 16 * DIM_K + kk * 32, DIM_K, lane);
            v16h a0 = lds_frag_a(feat + (trb + 0) * 16 * FEAT_LD + DIM_V + kk * 32, FEAT_LD, lane);
            v16h a1 = lds_frag_a(feat + (trb + 1) * 16 * FEAT_LD + DIM_V + kk * 32, FEAT_LD, lane);
            acc0 = wmma_f16(a0, b, acc0);
            acc1 = wmma_f16(a1, b, acc1);
        }
        const int n = tc * 16 + (lane & 15);
        const int m0 = (trb + 0) * 16 + (lane >> 4) * 8;
        const int m1 = (trb + 1) * 16 + (lane >> 4) * 8;
        #pragma unroll
        for (int r = 0; r < 8; ++r) {
            logits[(m0 + r) * MEM_PAD + n] = acc0[r];
            logits[(m1 + r) * MEM_PAD + n] = acc1[r];
        }
    }
    __syncthreads();

    // ---- phase 3: f32 softmax over the 50 valid slots
    if (tid < TILE_TOK) {
        float* row = logits + tid * MEM_PAD;
        float mx = -1e30f;
        for (int m = 0; m < 50; ++m) mx = fmaxf(mx, row[m]);
        float s = 0.0f;
        for (int m = 0; m < 50; ++m) { float e = __expf(row[m] - mx); row[m] = e; s += e; }
        const float inv = 1.0f / s;
        _Float16* crow = corr + tid * CORR_LD;
        for (int m = 0; m < 50; ++m)  crow[m] = (_Float16)(row[m] * inv);
        for (int m = 50; m < 64; ++m) crow[m] = (_Float16)0.0f;
    }
    __syncthreads();

    // ---- phase 4: read[64x256] = corr[64x64] @ vmT^T -> feat[:,0:256]
    // 64 tiles (4tr x 16tc); wave handles tc = wave*2+t, all 4 tr share B.
    for (int t = 0; t < 2; ++t) {
        const int tc = wave * 2 + t;                  // 0..15
        v8f acc[4] = {{}, {}, {}, {}};
        #pragma unroll
        for (int kk = 0; kk < 2; ++kk) {
            v16h b = glb_frag_b(vmt16 + tc * 16 * MEM_PAD + kk * 32, MEM_PAD, lane);
            #pragma unroll
            for (int tr = 0; tr < 4; ++tr) {
                v16h a = lds_frag_a(corr + tr * 16 * CORR_LD + kk * 32, CORR_LD, lane);
                acc[tr] = wmma_f16(a, b, acc[tr]);
            }
        }
        const int n = tc * 16 + (lane & 15);
        #pragma unroll
        for (int tr = 0; tr < 4; ++tr) {
            const int mbase = tr * 16 + (lane >> 4) * 8;
            #pragma unroll
            for (int r = 0; r < 8; ++r)
                feat[(mbase + r) * FEAT_LD + n] = (_Float16)acc[tr][r];
        }
    }
    __syncthreads();

    // ---- phase 5: state = relu(feat @ W_f^T + b_f), fused dot with W_p.
    // 128 tiles (4tr x 32tc), K=384; wave handles tc = wave*4+t; B fragments
    // loaded once per (tc,kk) and reused across the 4 tr tiles.
    for (int t = 0; t < 4; ++t) {
        const int tc = wave * 4 + t;                  // 0..31
        v8f acc[4] = {{}, {}, {}, {}};
        for (int kk = 0; kk < 12; ++kk) {
            v16h b = glb_frag_b(wf16 + (size_t)(tc * 16) * DIM_F + kk * 32, DIM_F, lane);
            #pragma unroll
            for (int tr = 0; tr < 4; ++tr) {
                v16h a = lds_frag_a(feat + tr * 16 * FEAT_LD + kk * 32, FEAT_LD, lane);
                acc[tr] = wmma_f16(a, b, acc[tr]);
            }
        }
        const int n  = tc * 16 + (lane & 15);
        const float bf = bfs[n];
        const float wp = wps[n];
        #pragma unroll
        for (int tr = 0; tr < 4; ++tr) {
            const int mbase = tr * 16 + (lane >> 4) * 8;
            #pragma unroll
            for (int r = 0; r < 8; ++r) {
                float v = acc[tr][r] + bf;
                v = fmaxf(v, 0.0f);
                float c = v * wp;
                // sum across the 16 lanes holding the same token
                c += __shfl_xor(c, 1, 16);
                c += __shfl_xor(c, 2, 16);
                c += __shfl_xor(c, 4, 16);
                c += __shfl_xor(c, 8, 16);
                if ((lane & 15) == 0) atomicAdd(&pacc[mbase + r], c);
            }
        }
    }
    __syncthreads();

    // ---- phase 6: sigmoid
    if (tid < TILE_TOK) {
        const float z = pacc[tid] + b_p[0];
        out[tok0 + tid] = 1.0f / (1.0f + __expf(-z));
    }
}

// ---------------------------------------------------------------------------
extern "C" void kernel_launch(void* const* d_in, const int* in_sizes, int n_in,
                              void* d_out, int out_size, void* d_ws, size_t ws_size,
                              hipStream_t stream) {
    (void)in_sizes; (void)n_in; (void)ws_size;
    const int*   q       = (const int*)  d_in[0];
    // d_in[1] = a (unused by the forward math)
    const float* key_mem = (const float*)d_in[2];
    const float* val_mem = (const float*)d_in[3];
    const float* embed   = (const float*)d_in[4];
    // d_in[5..8] = W_erase/b_erase/W_add/b_add (unused in forward)
    const float* W_f     = (const float*)d_in[9];
    const float* b_f     = (const float*)d_in[10];
    const float* W_p     = (const float*)d_in[11];
    const float* b_p     = (const float*)d_in[12];
    float*       out     = (float*)d_out;
    _Float16*    ws      = (_Float16*)d_ws;

    const int prep_elems = DIM_H * DIM_F + MEM_PAD * DIM_K + DIM_V * MEM_PAD;
    dkvmn_prep<<<(prep_elems + THREADS - 1) / THREADS, THREADS, 0, stream>>>(
        W_f, key_mem, val_mem, ws);

    const int n_tok  = out_size;                  // 128 * 2048
    const int blocks = (n_tok + TILE_TOK - 1) / TILE_TOK;
    dkvmn_main<<<blocks, THREADS, SM_BYTES, stream>>>(
        q, embed, b_f, W_p, b_p, ws, out);
}
